// FastWeightsAttention_3831110828605
// MI455X (gfx1250) — matrix operand order repaired and checked by reading, not compiled, not run on hardware
//
#include <hip/hip_runtime.h>
#include <hip/hip_bf16.h>
#include <stdint.h>

// ---------------------------------------------------------------------------
// FastWeightsAttention (DPFP linear attention w/ delta rule) for gfx1250.
//   S=4096, B=4, D=1024, H=16, dk=64, NU=1, d_dot=128
// Pipeline:
//   0) split_kernel : x -> xhi/xlo (row-major bf16 hi/lo, split once)
//      wfrag_kernel : Wq/Wk/Wv/Wo -> WMMA-B-fragment-layout bf16 hi/lo
//   1) gemm_wmma    : qraw/kraw/vbuf = x@W*      (split-bf16, 6 WMMA / chunk)
//   2) beta_kernel  : blog = x@Wb (logits; sigmoid in scan)
//   3) dpfp_kernel  : qphi/kphi = dpfp(qraw/kraw)
//   4) scan_kernel  : delta-rule recurrence, one block per (b,h)
//   5) split + gemm : out = y@Wo + bo
// ---------------------------------------------------------------------------

typedef __attribute__((ext_vector_type(16))) __bf16     v16bf;
typedef __attribute__((ext_vector_type(8)))  __bf16     v8bf;
typedef __attribute__((ext_vector_type(4)))  __bf16     v4bf;
typedef __attribute__((ext_vector_type(8)))  float      v8f;
typedef __attribute__((ext_vector_type(4)))  unsigned   v4u;
typedef __attribute__((ext_vector_type(4)))  int        v4i;
typedef __attribute__((ext_vector_type(8)))  int        v8i;

#define SEQ   4096
#define BATCH 4
#define DIM   1024
#define HEADS 16
#define DK    64
#define DDOT  128
#define KDIM  1024   // GEMM K == DIM (fixed; LDS panels sized for it)

// ---------------- TDM: 2-D tile load (rows x dim0 elements) into LDS --------
__device__ inline void tdm_load_2d(const void* gptr, unsigned lds_addr,
                                   unsigned dim0, unsigned tile_rows,
                                   unsigned dim1_total, unsigned ds_code) {
#if __has_builtin(__builtin_amdgcn_tensor_load_to_lds)
  unsigned long long gaddr = (unsigned long long)(uintptr_t)gptr;
  // group0: count=1 | lds_addr<<32 ; global_addr(57b) | type=2<<62
  unsigned long long q0 = 1ull | ((unsigned long long)lds_addr << 32);
  unsigned long long q1 = (gaddr & ((1ull << 57) - 1)) | (2ull << 62);
  v4u g0 = { (unsigned)q0, (unsigned)(q0 >> 32),
             (unsigned)q1, (unsigned)(q1 >> 32) };
  // group1: data_size@16 | tensor_dim0@48 | tensor_dim1@80 | tile_dim0@112
  //         tile_dim1@128 | tensor_dim0_stride@160
  unsigned long long r0 = ((unsigned long long)ds_code << 16)
                        | ((unsigned long long)(dim0 & 0xFFFF) << 48);
  unsigned long long r1 = (unsigned long long)(dim0 >> 16)
                        | ((unsigned long long)dim1_total << 16)
                        | ((unsigned long long)(dim0 & 0xFFFF) << 48);
  unsigned long long r2 = (unsigned long long)tile_rows
                        | ((unsigned long long)dim0 << 32);
  unsigned long long r3 = 0ull;
  v8i g1 = { (int)r0, (int)(r0 >> 32), (int)r1, (int)(r1 >> 32),
             (int)r2, (int)(r2 >> 32), (int)r3, (int)(r3 >> 32) };
  v4i z2 = {0, 0, 0, 0};
  v4i z3 = {0, 0, 0, 0};
#if defined(__clang_major__) && (__clang_major__ >= 23)
  v8i z4 = {0, 0, 0, 0, 0, 0, 0, 0};
  __builtin_amdgcn_tensor_load_to_lds(g0, g1, z2, z3, z4, 0);
#else
  __builtin_amdgcn_tensor_load_to_lds(g0, g1, z2, z3, 0);
#endif
#endif
}

// ---------------- prep: fp32 -> row-major bf16 hi/lo ------------------------
__global__ __launch_bounds__(256) void split_kernel(const float* __restrict__ in,
                                                    __bf16* __restrict__ hi,
                                                    __bf16* __restrict__ lo,
                                                    size_t n4) {
  size_t i = (size_t)blockIdx.x * blockDim.x + threadIdx.x;
  if (i >= n4) return;
  float4 v = ((const float4*)in)[i];
  __bf16 h0 = (__bf16)v.x, h1 = (__bf16)v.y, h2 = (__bf16)v.z, h3 = (__bf16)v.w;
  v4bf hv = { h0, h1, h2, h3 };
  v4bf lv = { (__bf16)(v.x - (float)h0), (__bf16)(v.y - (float)h1),
              (__bf16)(v.z - (float)h2), (__bf16)(v.w - (float)h3) };
  ((v4bf*)hi)[i] = hv;
  ((v4bf*)lo)[i] = lv;
}

// ---------------- prep: weight [K,N] fp32 -> B-fragment bf16 hi/lo ----------
// dword id = ((c*N + n)*2 + hlf)*8 + v  holds pair { B[k][n], B[k+1][n] },
// k = c*32 + hlf*8 + 2(v&3) + 16(v>>2).  GEMM then loads a whole v16bf fragment contiguously.
__device__ inline unsigned pack_bf16(__bf16 a, __bf16 b) {
  union { __bf16 h[2]; unsigned u; } t;
  t.h[0] = a; t.h[1] = b;
  return t.u;
}

__global__ __launch_bounds__(256) void wfrag_kernel(const float* __restrict__ W,
                                                    unsigned* __restrict__ fhi,
                                                    unsigned* __restrict__ flo,
                                                    int K, int N) {
  int id = blockIdx.x * blockDim.x + threadIdx.x;
  if (id >= (K / 2) * N) return;
  int v    = id & 7;
  int hlf  = (id >> 3) & 1;
  int rest = id >> 4;
  int n    = rest % N;
  int c    = rest / N;
  int k    = c * 32 + hlf * 8 + (v & 3) * 2 + (v >> 2) * 16;
  float b0 = W[(size_t)k * N + n];
  float b1 = W[(size_t)(k + 1) * N + n];
  __bf16 h0 = (__bf16)b0, h1 = (__bf16)b1;
  fhi[id] = pack_bf16(h0, h1);
  flo[id] = pack_bf16((__bf16)(b0 - (float)h0), (__bf16)(b1 - (float)h1));
}

// ---------------- GEMM: C[M,N] = A@B (+bias), split-bf16 via WMMA -----------
// Block: 256 thr (8 waves), tile 32(M) x 128(N); wave owns two 16x16 accs.
// A hi/lo panels (32 x K bf16 each, 64KB) staged via TDM. B fragments read
// directly from prepacked layout (two b128 per chunk per matrix).
__device__ inline v16bf ldfrag(const __bf16* p) {
  v8bf x0 = *(const v8bf*)p;
  v8bf x1 = *(const v8bf*)(p + 16);
  return __builtin_shufflevector(x0, x1, 0, 1, 2, 3, 4, 5, 6, 7,
                                 8, 9, 10, 11, 12, 13, 14, 15);
}

__global__ __launch_bounds__(256) void gemm_wmma(const __bf16* __restrict__ Ahi,
                                                 const __bf16* __restrict__ Alo,
                                                 const unsigned* __restrict__ Bhi,
                                                 const unsigned* __restrict__ Blo,
                                                 const float* __restrict__ bias,
                                                 float* __restrict__ C,
                                                 int M, int N, int K) {
  __shared__ __bf16 sAhi[32 * KDIM];   // 64 KB
  __shared__ __bf16 sAlo[32 * KDIM];   // 64 KB

  const int tileM = blockIdx.y * 32;
  const int wave  = threadIdx.x >> 5;
  const int lane  = threadIdx.x & 31;
  const int tileN = blockIdx.x * 128 + wave * 16;
  const int m     = lane & 15;
  const int hlf   = lane >> 4;
  const int n     = tileN + m;

#if __has_builtin(__builtin_amdgcn_tensor_load_to_lds)
  if ((threadIdx.x >> 5) == 0) {
    tdm_load_2d(Ahi + (size_t)tileM * K, (unsigned)(uintptr_t)(&sAhi[0]),
                (unsigned)K, 32u, (unsigned)M, 1u /* 2-byte elems */);
    tdm_load_2d(Alo + (size_t)tileM * K, (unsigned)(uintptr_t)(&sAlo[0]),
                (unsigned)K, 32u, (unsigned)M, 1u);
    __builtin_amdgcn_s_wait_tensorcnt(0);
  }
#else
  {
    const unsigned* src0 = (const unsigned*)(Ahi + (size_t)tileM * K);
    const unsigned* src1 = (const unsigned*)(Alo + (size_t)tileM * K);
    for (int i = threadIdx.x; i < 32 * KDIM / 2; i += 256) {
      ((unsigned*)sAhi)[i] = src0[i];
      ((unsigned*)sAlo)[i] = src1[i];
    }
  }
#endif
  __syncthreads();

  v8f acc0 = {0.f, 0.f, 0.f, 0.f, 0.f, 0.f, 0.f, 0.f};
  v8f acc1 = {0.f, 0.f, 0.f, 0.f, 0.f, 0.f, 0.f, 0.f};

  for (int c = 0; c < K / 32; ++c) {
    const int k0 = c * 32;
    // B fragments: prepacked, 32B contiguous & aligned per (chunk, col, half)
    const size_t bidx = ((size_t)c * N + n) * 16 + (size_t)hlf * 8;
    v16bf bh = *(const v16bf*)(Bhi + bidx);
    v16bf bl = *(const v16bf*)(Blo + bidx);
    // A fragments: 16x32 bf16 layout = 8 bf16 @ (k0+8h) and 8 bf16 @ (+16)
    const int ao = k0 + (hlf << 3);
    v16bf a0h = ldfrag(&sAhi[(size_t)m * KDIM + ao]);
    v16bf a0l = ldfrag(&sAlo[(size_t)m * KDIM + ao]);
    v16bf a1h = ldfrag(&sAhi[(size_t)(m + 16) * KDIM + ao]);
    v16bf a1l = ldfrag(&sAlo[(size_t)(m + 16) * KDIM + ao]);
    // fp32 emulation: a*b ~= ah*bh + ah*bl + al*bh (f32 accumulate)
    acc0 = __builtin_amdgcn_wmma_f32_16x16x32_bf16(false, a0h, false, bh,
                                                   (short)0, acc0, false, false);
    acc1 = __builtin_amdgcn_wmma_f32_16x16x32_bf16(false, a1h, false, bh,
                                                   (short)0, acc1, false, false);
    acc0 = __builtin_amdgcn_wmma_f32_16x16x32_bf16(false, a0h, false, bl,
                                                   (short)0, acc0, false, false);
    acc1 = __builtin_amdgcn_wmma_f32_16x16x32_bf16(false, a1h, false, bl,
                                                   (short)0, acc1, false, false);
    acc0 = __builtin_amdgcn_wmma_f32_16x16x32_bf16(false, a0l, false, bh,
                                                   (short)0, acc0, false, false);
    acc1 = __builtin_amdgcn_wmma_f32_16x16x32_bf16(false, a1l, false, bh,
                                                   (short)0, acc1, false, false);
  }

  const float bv = bias ? bias[n] : 0.f;
#pragma unroll
  for (int r = 0; r < 8; ++r) {   // C layout: VGPR r -> M = r + 8*half
    C[(size_t)(tileM + r + (hlf << 3)) * N + n]      = acc0[r] + bv;
    C[(size_t)(tileM + 16 + r + (hlf << 3)) * N + n] = acc1[r] + bv;
  }
}

// ======================= beta logits: x[M,1024] @ Wb[1024,16] ===============
__global__ __launch_bounds__(256) void beta_kernel(const float* __restrict__ x,
                                                   const float* __restrict__ Wb,
                                                   float* __restrict__ blog,
                                                   int Mrows) {
  int idx = blockIdx.x * blockDim.x + threadIdx.x;
  if (idx >= Mrows * HEADS) return;
  int row = idx >> 4, h = idx & 15;
  const float* xr = x + (size_t)row * DIM;
  float acc = 0.f;
  for (int k = 0; k < DIM; ++k) acc = fmaf(xr[k], Wb[k * HEADS + h], acc);
  blog[idx] = acc;
}

// ======================= DPFP: [rows,64] -> normalized [rows,128] ===========
__device__ inline float dpfp_x(const float* __restrict__ r, int i) {
  float v = (i < DK) ? r[i] : -r[i - DK];
  return v > 0.f ? v : 0.f;
}

__global__ __launch_bounds__(256) void dpfp_kernel(const float* __restrict__ raw,
                                                   float* __restrict__ phi,
                                                   int nrows) {
  int w = (blockIdx.x * blockDim.x + threadIdx.x) >> 5;   // one wave per row
  int lane = threadIdx.x & 31;
  if (w >= nrows) return;
  const float* r = raw + (size_t)w * DK;
  float* p = phi + (size_t)w * DDOT;
  float ph[4];
  float s = 0.f;
#pragma unroll
  for (int j = 0; j < 4; ++j) {
    int i = lane * 4 + j;
    float xi = dpfp_x(r, i);
    float xp = dpfp_x(r, (i + DDOT - 1) & (DDOT - 1));  // roll by 1
    ph[j] = xi * xp;
    s += ph[j];
  }
#pragma unroll
  for (int off = 16; off > 0; off >>= 1) s += __shfl_xor(s, off, 32);
  float inv = 1.f / (s + 1e-6f);
#pragma unroll
  for (int j = 0; j < 4; ++j) p[lane * 4 + j] = ph[j] * inv;
}

// ======================= delta-rule scan, one block per (b,h) ===============
// W state [64 x 128] in registers across 256 threads. Two barriers per step:
//   y = W_new q = W_old q + delta * (k . q)  -> one reduction covers all.
__global__ __launch_bounds__(256) void scan_kernel(const float* __restrict__ kphi,
                                                   const float* __restrict__ qphi,
                                                   const float* __restrict__ vbuf,
                                                   const float* __restrict__ blog,
                                                   float* __restrict__ ybuf) {
  const int bh = blockIdx.x;
  const int b = bh / HEADS, h = bh % HEADS;
  const int t = threadIdx.x;
  const int j = t & 63;
  const int qr = t >> 6;
  const int kbase = qr * 32;

  __shared__ float ksh[2][DDOT], qsh[2][DDOT];
  __shared__ float red1[256], red2[256], kqred[4], dsh[64];

  float W[32];
#pragma unroll
  for (int c = 0; c < 32; ++c) W[c] = 0.f;

  {  // preload step 0
    size_t nb = (size_t)(0 * BATCH + b) * HEADS + h;
    if (t < DDOT) ksh[0][t] = kphi[nb * DDOT + t];
    else          qsh[0][t - DDOT] = qphi[nb * DDOT + (t - DDOT)];
  }
  __syncthreads();

  for (int s = 0; s < SEQ; ++s) {
    const int cur = s & 1, nxt = cur ^ 1;
    const float* kc = ksh[cur];
    const float* qc = qsh[cur];

    float p1 = 0.f, p2 = 0.f;
#pragma unroll
    for (int c = 0; c < 32; ++c) {
      float kv = kc[kbase + c], qv = qc[kbase + c];
      p1 = fmaf(W[c], kv, p1);
      p2 = fmaf(W[c], qv, p2);
    }
    red1[t] = p1;
    red2[t] = p2;
    if (j == 0) {
      float pk = 0.f;
#pragma unroll
      for (int c = 0; c < 32; ++c) pk = fmaf(kc[kbase + c], qc[kbase + c], pk);
      kqred[qr] = pk;
    }
    if (s + 1 < SEQ) {  // double-buffered prefetch of next step's k/q
      size_t nb = ((size_t)(s + 1) * BATCH + b) * HEADS + h;
      if (t < DDOT) ksh[nxt][t] = kphi[nb * DDOT + t];
      else          qsh[nxt][t - DDOT] = qphi[nb * DDOT + (t - DDOT)];
    }
    __syncthreads();

    if (qr == 0) {  // t < 64: j = t
      size_t base = ((size_t)s * BATCH + b) * HEADS + h;
      float vbar = red1[j] + red1[64 + j] + red1[128 + j] + red1[192 + j];
      float yold = red2[j] + red2[64 + j] + red2[128 + j] + red2[192 + j];
      float kq   = kqred[0] + kqred[1] + kqred[2] + kqred[3];
      float beta = 1.f / (1.f + __expf(-blog[base]));
      float d    = beta * (vbuf[base * DK + j] - vbar);
      dsh[j] = d;
      ybuf[base * DK + j] = yold + d * kq;
    }
    __syncthreads();

    float d = dsh[j];
#pragma unroll
    for (int c = 0; c < 32; ++c) W[c] = fmaf(d, kc[kbase + c], W[c]);
  }
}

// ======================= host-side launch ===================================
extern "C" void kernel_launch(void* const* d_in, const int* in_sizes, int n_in,
                              void* d_out, int out_size, void* d_ws, size_t ws_size,
                              hipStream_t stream) {
  (void)in_sizes; (void)n_in; (void)out_size; (void)ws_size;
  const float* x  = (const float*)d_in[0];
  const float* Wq = (const float*)d_in[1];
  const float* Wk = (const float*)d_in[2];
  const float* Wv = (const float*)d_in[3];
  const float* Wb = (const float*)d_in[4];
  const float* Wo = (const float*)d_in[5];
  const float* bo = (const float*)d_in[6];
  float* out = (float*)d_out;

  const int M = SEQ * BATCH;                                  // 16384
  const size_t szMD    = (size_t)M * DIM * sizeof(float);     // 64 MB
  const size_t szMDbf  = (size_t)M * DIM * sizeof(__bf16);    // 32 MB
  const size_t szWfrag = (size_t)(KDIM / 2) * DIM * sizeof(unsigned); // 2 MB
  const size_t szPhi   = (size_t)M * HEADS * DDOT * sizeof(float);
  const size_t szBlog  = (size_t)M * HEADS * sizeof(float);

  char* ws = (char*)d_ws;
  size_t off = 0;
  __bf16* xhi = (__bf16*)(ws + off); off += szMDbf;
  __bf16* xlo = (__bf16*)(ws + off); off += szMDbf;
  unsigned* wf[8];                       // {q,k,v,o} x {hi,lo}
  for (int i = 0; i < 8; ++i) { wf[i] = (unsigned*)(ws + off); off += szWfrag; }
  float* qraw = (float*)(ws + off); off += szMD;
  float* kraw = (float*)(ws + off); off += szMD;
  float* vbuf = (float*)(ws + off); off += szMD;
  float* blog = (float*)(ws + off); off += szBlog;
  float* qphi = (float*)(ws + off); off += szPhi;
  float* kphi = (float*)(ws + off); off += szPhi;
  float* ybuf = qraw;                    // qraw dead after dpfp
  __bf16* yhi = xhi;                     // x splits dead after projections
  __bf16* ylo = xlo;

  const size_t n4x = (size_t)M * DIM / 4;
  const int nWfrag = (KDIM / 2) * DIM;
  dim3 gblk(256);
  dim3 ggrid(DIM / 128, M / 32);         // (8, 512)

  // prep
  split_kernel<<<(unsigned)((n4x + 255) / 256), gblk, 0, stream>>>(x, xhi, xlo, n4x);
  wfrag_kernel<<<(nWfrag + 255) / 256, gblk, 0, stream>>>(Wq, wf[0], wf[1], KDIM, DIM);
  wfrag_kernel<<<(nWfrag + 255) / 256, gblk, 0, stream>>>(Wk, wf[2], wf[3], KDIM, DIM);
  wfrag_kernel<<<(nWfrag + 255) / 256, gblk, 0, stream>>>(Wv, wf[4], wf[5], KDIM, DIM);
  wfrag_kernel<<<(nWfrag + 255) / 256, gblk, 0, stream>>>(Wo, wf[6], wf[7], KDIM, DIM);

  // projections
  gemm_wmma<<<ggrid, gblk, 0, stream>>>(xhi, xlo, wf[0], wf[1], nullptr, qraw, M, DIM, KDIM);
  gemm_wmma<<<ggrid, gblk, 0, stream>>>(xhi, xlo, wf[2], wf[3], nullptr, kraw, M, DIM, KDIM);
  gemm_wmma<<<ggrid, gblk, 0, stream>>>(xhi, xlo, wf[4], wf[5], nullptr, vbuf, M, DIM, KDIM);
  beta_kernel<<<(M * HEADS + 255) / 256, gblk, 0, stream>>>(x, Wb, blog, M);

  // dpfp
  const int nrows = M * HEADS;
  dpfp_kernel<<<(nrows * 32 + 255) / 256, gblk, 0, stream>>>(qraw, qphi, nrows);
  dpfp_kernel<<<(nrows * 32 + 255) / 256, gblk, 0, stream>>>(kraw, kphi, nrows);

  // recurrence
  scan_kernel<<<BATCH * HEADS, gblk, 0, stream>>>(kphi, qphi, vbuf, blog, ybuf);

  // output projection
  split_kernel<<<(unsigned)((n4x + 255) / 256), gblk, 0, stream>>>(ybuf, yhi, ylo, n4x);
  gemm_wmma<<<ggrid, gblk, 0, stream>>>(yhi, ylo, wf[6], wf[7], bo, out, M, DIM, KDIM);
}
